// MambaClassifier_37984690766272
// MI455X (gfx1250) — compile-verified, matrix operand order checked by reading, not run on hardware
//
#include <hip/hip_runtime.h>
#include <hip/hip_bf16.h>

typedef __attribute__((ext_vector_type(16))) __bf16 v16bf;
typedef __attribute__((ext_vector_type(8)))  __bf16 v8bf;
typedef __attribute__((ext_vector_type(8)))  float  v8f;

#define HDIM 768
#define IDIM 1536
#define NSTATE 16
#define KCONV 4
#define RRANK 48
#define DBCW 80      /* R + 2N */
#define NLAYERS 32
#define BB 8
#define LL 1024
#define MROWS (BB*LL)
#define CCLS 16
#define EPSV 1e-5f

static __device__ __forceinline__ v8f zero_v8f() {
  v8f z;
#pragma unroll
  for (int k = 0; k < 8; k++) z[k] = 0.0f;
  return z;
}
static __device__ __forceinline__ v8bf zero_v8bf() {
  v8bf z;
#pragma unroll
  for (int k = 0; k < 8; k++) z[k] = (__bf16)0.0f;
  return z;
}

/* ---------------- fp32 -> bf16 weight conversion ---------------- */
__global__ void cvt_f32_bf16(const float* __restrict__ src, __bf16* __restrict__ dst, long n) {
  long i = (long)blockIdx.x * blockDim.x + threadIdx.x;
  long stride = (long)gridDim.x * blockDim.x;
  for (; i < n; i += stride) dst[i] = (__bf16)src[i];
}

/* ---------------- embedding gather + mask ---------------- */
__global__ void embed_kernel(const int* __restrict__ tokens, const float* __restrict__ emb,
                             float* __restrict__ h, float* __restrict__ mask) {
  int row = blockIdx.x;
  int tok = tokens[row];
  const float* e = emb + (size_t)tok * HDIM;
  float* ho = h + (size_t)row * HDIM;
  for (int c = threadIdx.x; c < HDIM; c += blockDim.x) ho[c] = e[c];
  if (threadIdx.x == 0) mask[row] = (tok != 0) ? 1.0f : 0.0f;
}

/* ---------------- block reduction helper (256 threads) ---------------- */
static __device__ __forceinline__ float block_reduce_sum(float v, float* red) {
#pragma unroll
  for (int o = 16; o > 0; o >>= 1) v += __shfl_down(v, o, 32);
  int wid = threadIdx.x >> 5, lane = threadIdx.x & 31;
  __syncthreads();
  if (lane == 0) red[wid] = v;
  __syncthreads();
  float tot = 0.0f;
#pragma unroll
  for (int k = 0; k < 8; k++) tot += red[k];
  return tot;
}

/* ---------------- per-layer RMSNorm -> bf16 x ---------------- */
__global__ __launch_bounds__(256) void rmsnorm_kernel(const float* __restrict__ h,
                                                      const float* __restrict__ w,
                                                      __bf16* __restrict__ xb) {
  __shared__ float red[8];
  int row = blockIdx.x;
  const float* hr = h + (size_t)row * HDIM;
  int t = threadIdx.x;
  float v0 = hr[t], v1 = hr[t + 256], v2 = hr[t + 512];
  float tot = block_reduce_sum(v0 * v0 + v1 * v1 + v2 * v2, red);
  float sc = rsqrtf(tot / (float)HDIM + EPSV);
  __bf16* xr = xb + (size_t)row * HDIM;
  xr[t]       = (__bf16)(v0 * sc * w[t]);
  xr[t + 256] = (__bf16)(v1 * sc * w[t + 256]);
  xr[t + 512] = (__bf16)(v2 * sc * w[t + 512]);
}

/* ---------------- bf16 WMMA GEMM: C[M,N] (+)= A[M,K] * Bw[N,K]^T ----------------
 * Block: 256 thr (8 waves). Block tile 128(M) x 128(N), K chunks of 32,
 * double-buffered LDS. Wave grid 2(M) x 4(N); wave tile 64x32 = 4x2 WMMA frags.
 * Requires: M % 128 == 0, K % 32 == 0. NGUARD: bounds-check columns vs N.  */
#define TM 128
#define TN 128
#define TK 32
#define LDP 40 /* padded LDS row (bf16): 80B rows keep 16B alignment, skew banks */

template <bool NGUARD>
__global__ __launch_bounds__(256) void gemm_bf16_wmma(const __bf16* __restrict__ A,
                                                      const __bf16* __restrict__ Bw,
                                                      float* __restrict__ C,
                                                      int M, int N, int Kd, int accFlag) {
  __shared__ __align__(16) __bf16 As[2][TM * LDP];
  __shared__ __align__(16) __bf16 Bs[2][TN * LDP];
  const int t = threadIdx.x;
  const int lane = t & 31, wid = t >> 5;
  const int wm = wid & 1;   /* 2 M-halves of 64 rows   */
  const int wn = wid >> 1;  /* 4 N-quarters of 32 cols */
  const int half = lane >> 4, r16 = lane & 15;
  const int mBase = blockIdx.y * TM;
  const int nBase = blockIdx.x * TN;

  v8f acc[4][2];
#pragma unroll
  for (int i = 0; i < 4; i++)
#pragma unroll
    for (int j = 0; j < 2; j++) acc[i][j] = zero_v8f();

  /* tile staging: each thread owns 2 rows of A-tile and 2 rows of B-tile,
   * one 16B chunk (8 bf16) per row: rows t>>2 and 64+(t>>2), chunk (t&3)*8 */
  const int srow = t >> 2;
  const int c8 = (t & 3) * 8;
  const size_t aOff0 = (size_t)(mBase + srow) * Kd + c8;
  const size_t aOff1 = (size_t)(mBase + srow + 64) * Kd + c8;
  const size_t bOff0 = (size_t)(nBase + srow) * Kd + c8;
  const size_t bOff1 = (size_t)(nBase + srow + 64) * Kd + c8;
  const bool bok0 = !NGUARD || (nBase + srow) < N;
  const bool bok1 = !NGUARD || (nBase + srow + 64) < N;

  v8bf rA0, rA1, rB0, rB1;
  auto loadTiles = [&](int k0) {
    rA0 = *(const v8bf*)(A + aOff0 + k0);
    rA1 = *(const v8bf*)(A + aOff1 + k0);
    if (NGUARD) {
      rB0 = bok0 ? *(const v8bf*)(Bw + bOff0 + k0) : zero_v8bf();
      rB1 = bok1 ? *(const v8bf*)(Bw + bOff1 + k0) : zero_v8bf();
    } else {
      rB0 = *(const v8bf*)(Bw + bOff0 + k0);
      rB1 = *(const v8bf*)(Bw + bOff1 + k0);
    }
  };
  auto stage = [&](int buf) {
    *(v8bf*)(&As[buf][srow * LDP + c8])        = rA0;
    *(v8bf*)(&As[buf][(srow + 64) * LDP + c8]) = rA1;
    *(v8bf*)(&Bs[buf][srow * LDP + c8])        = rB0;
    *(v8bf*)(&Bs[buf][(srow + 64) * LDP + c8]) = rB1;
  };

  loadTiles(0);
  stage(0);
  __syncthreads();

  const int nch = Kd / TK;
  for (int c = 0; c < nch; c++) {
    const int cur = c & 1;
    const bool more = (c + 1) < nch;
    if (more) {
      loadTiles((c + 1) * TK); /* overlap with WMMA below */
      if (c + 2 < nch) {       /* gfx1250 global_prefetch_b8 for chunk c+2 */
        __builtin_prefetch(A + aOff0 + (size_t)(c + 2) * TK, 0, 0);
        __builtin_prefetch(Bw + bOff0 + (size_t)(c + 2) * TK, 0, 0);
      }
    }

    /* fragment loads per ISA 16-bit A layout:
     * lane<16 holds K{0..7,16..23}, lane>=16 holds K{8..15,24..31} */
    union Frag { v16bf v; v8bf h2[2]; };
    Frag fa[4], fb[2];
#pragma unroll
    for (int i = 0; i < 4; i++) {
      const __bf16* p = &As[cur][(wm * 64 + i * 16 + r16) * LDP];
      fa[i].h2[0] = *(const v8bf*)(p + half * 8);
      fa[i].h2[1] = *(const v8bf*)(p + 16 + half * 8);
    }
#pragma unroll
    for (int j = 0; j < 2; j++) {
      const __bf16* p = &Bs[cur][(wn * 32 + j * 16 + r16) * LDP];
      fb[j].h2[0] = *(const v8bf*)(p + half * 8);
      fb[j].h2[1] = *(const v8bf*)(p + 16 + half * 8);
    }
#pragma unroll
    for (int i = 0; i < 4; i++)
#pragma unroll
      for (int j = 0; j < 2; j++)
        acc[i][j] = __builtin_amdgcn_wmma_f32_16x16x32_bf16(
            false, fa[i].v, false, fb[j].v, (short)0, acc[i][j], false, false);

    if (more) {
      stage(cur ^ 1); /* other buffer: no race with current readers */
      __syncthreads();
    }
  }

  /* C/D layout: VGPR v -> M = v + 8*half, N = r16 */
#pragma unroll
  for (int i = 0; i < 4; i++)
#pragma unroll
    for (int j = 0; j < 2; j++) {
      int col = nBase + wn * 32 + j * 16 + r16;
      if (!NGUARD || col < N) {
#pragma unroll
        for (int v = 0; v < 8; v++) {
          int row = mBase + wm * 64 + i * 16 + half * 8 + v;
          size_t idx = (size_t)row * N + col;
          if (accFlag) C[idx] += acc[i][j][v];
          else C[idx] = acc[i][j][v];
        }
      }
    }
}

/* ---------------- causal depthwise conv (K=4) + silu + mask ---------------- */
__global__ void conv_silu_kernel(const float* __restrict__ proj, const float* __restrict__ mask,
                                 const float* __restrict__ cw, const float* __restrict__ cb,
                                 float* __restrict__ u2f, __bf16* __restrict__ u2b) {
  int tid = blockIdx.x * blockDim.x + threadIdx.x;
  if (tid >= BB * IDIM) return;
  int b = tid / IDIM, i = tid - b * IDIM;
  float c0 = cw[i * KCONV + 0], c1 = cw[i * KCONV + 1];
  float c2 = cw[i * KCONV + 2], c3 = cw[i * KCONV + 3];
  float bias = cb[i];
  float w0 = 0.f, w1 = 0.f, w2 = 0.f, w3 = 0.f;
  for (int l = 0; l < LL; l++) {
    int row = b * LL + l;
    float mk = mask[row];
    float u = proj[(size_t)row * (2 * IDIM) + i] * mk;
    w0 = w1; w1 = w2; w2 = w3; w3 = u;
    float cv = bias + w0 * c0 + w1 * c1 + w2 * c2 + w3 * c3;
    float s = cv / (1.0f + __expf(-cv)); /* silu */
    float u2 = s * mk;
    u2f[(size_t)row * IDIM + i] = u2;
    u2b[(size_t)row * IDIM + i] = (__bf16)u2;
  }
}

/* ---------------- dt = softplus(dtr @ dt_w^T + dt_b), K=48 scalar ---------------- */
__global__ void dt_kernel(const float* __restrict__ dbc, const float* __restrict__ dtw,
                          const float* __restrict__ dtb, float* __restrict__ dt) {
  long tid = (long)blockIdx.x * blockDim.x + threadIdx.x;
  if (tid >= (long)MROWS * IDIM) return;
  int row = (int)(tid / IDIM), i = (int)(tid - (long)row * IDIM);
  const float* dr = dbc + (size_t)row * DBCW;
  const float* wr = dtw + (size_t)i * RRANK;
  float s = dtb[i];
#pragma unroll 8
  for (int r = 0; r < RRANK; r++) s += dr[r] * wr[r];
  float sp = (s > 20.0f) ? s : log1pf(__expf(s));
  dt[(size_t)row * IDIM + i] = sp;
}

/* ---------------- selective scan: one lane per (b, channel) ---------------- */
__global__ void scan_kernel(const float* __restrict__ dbc, const float* __restrict__ dt,
                            const float* __restrict__ u2f, const float* __restrict__ proj,
                            const float* __restrict__ A_log, const float* __restrict__ Dp,
                            __bf16* __restrict__ yb) {
  int tid = blockIdx.x * blockDim.x + threadIdx.x;
  if (tid >= BB * IDIM) return;
  int b = tid / IDIM, i = tid - b * IDIM;
  float Ac[NSTATE], st[NSTATE];
#pragma unroll
  for (int n = 0; n < NSTATE; n++) {
    Ac[n] = -__expf(A_log[i * NSTATE + n]);
    st[n] = 0.0f;
  }
  float Dv = Dp[i];
  for (int l = 0; l < LL; l++) {
    int row = b * LL + l;
    float dtv = dt[(size_t)row * IDIM + i];
    float uv  = u2f[(size_t)row * IDIM + i];
    float dtu = dtv * uv;
    const float* bc = dbc + (size_t)row * DBCW;
    float y = 0.0f;
#pragma unroll
    for (int n = 0; n < NSTATE; n++) {
      float dA = __expf(dtv * Ac[n]);
      st[n] = dA * st[n] + dtu * bc[RRANK + n];
      y += st[n] * bc[RRANK + NSTATE + n];
    }
    float g = proj[(size_t)row * (2 * IDIM) + IDIM + i];
    float sg = g / (1.0f + __expf(-g));
    yb[(size_t)row * IDIM + i] = (__bf16)((y + uv * Dv) * sg);
  }
}

/* ---------------- final rmsnorm + mean pool + FC (one block per batch) ---------------- */
__global__ __launch_bounds__(256) void final_kernel(const float* __restrict__ h,
                                                    const float* __restrict__ nfw,
                                                    const float* __restrict__ fcw,
                                                    const float* __restrict__ fcb,
                                                    float* __restrict__ out) {
  __shared__ float red[8];
  __shared__ float pooled[HDIM];
  int b = blockIdx.x, t = threadIdx.x;
  float a0 = 0.f, a1 = 0.f, a2 = 0.f;
  float nf0 = nfw[t], nf1 = nfw[t + 256], nf2 = nfw[t + 512];
  for (int l = 0; l < LL; l++) {
    const float* hr = h + (size_t)(b * LL + l) * HDIM;
    float v0 = hr[t], v1 = hr[t + 256], v2 = hr[t + 512];
    float tot = block_reduce_sum(v0 * v0 + v1 * v1 + v2 * v2, red);
    float sc = rsqrtf(tot / (float)HDIM + EPSV);
    a0 += v0 * sc * nf0; a1 += v1 * sc * nf1; a2 += v2 * sc * nf2;
  }
  pooled[t] = a0 / (float)LL;
  pooled[t + 256] = a1 / (float)LL;
  pooled[t + 512] = a2 / (float)LL;
  __syncthreads();
  if (t < CCLS) {
    float s = fcb[t];
    for (int hh = 0; hh < HDIM; hh++) s += pooled[hh] * fcw[t * HDIM + hh];
    out[b * CCLS + t] = s;
  }
}

extern "C" void kernel_launch(void* const* d_in, const int* in_sizes, int n_in,
                              void* d_out, int out_size, void* d_ws, size_t ws_size,
                              hipStream_t stream) {
  const int*   tokens     = (const int*)d_in[0];
  const float* emb        = (const float*)d_in[1];
  const float* in_proj_w  = (const float*)d_in[2];
  const float* conv_w     = (const float*)d_in[3];
  const float* conv_b     = (const float*)d_in[4];
  const float* x_proj_w   = (const float*)d_in[5];
  const float* dt_w       = (const float*)d_in[6];
  const float* dt_b       = (const float*)d_in[7];
  const float* A_log      = (const float*)d_in[8];
  const float* Dp         = (const float*)d_in[9];
  const float* out_proj_w = (const float*)d_in[10];
  const float* norm_w     = (const float*)d_in[11];
  const float* norm_f_w   = (const float*)d_in[12];
  const float* fc_w       = (const float*)d_in[13];
  const float* fc_b       = (const float*)d_in[14];
  float* out = (float*)d_out;
  (void)in_sizes; (void)n_in; (void)out_size; (void)ws_size;

  char* ws = (char*)d_ws;
  size_t off = 0;
  auto carve = [&](size_t bytes) -> char* {
    char* p = ws + off;
    off += (bytes + 255) & ~(size_t)255;
    return p;
  };
  __bf16* w_in_b  = (__bf16*)carve((size_t)NLAYERS * 2 * IDIM * HDIM * sizeof(__bf16));
  __bf16* w_out_b = (__bf16*)carve((size_t)NLAYERS * HDIM * IDIM * sizeof(__bf16));
  __bf16* w_xp_b  = (__bf16*)carve((size_t)NLAYERS * DBCW * IDIM * sizeof(__bf16));
  float*  hbuf    = (float*)carve((size_t)MROWS * HDIM * sizeof(float));
  float*  maskb   = (float*)carve((size_t)MROWS * sizeof(float));
  __bf16* xb      = (__bf16*)carve((size_t)MROWS * HDIM * sizeof(__bf16));
  float*  projb   = (float*)carve((size_t)MROWS * 2 * IDIM * sizeof(float));
  float*  u2f     = (float*)carve((size_t)MROWS * IDIM * sizeof(float));
  __bf16* u2b     = (__bf16*)carve((size_t)MROWS * IDIM * sizeof(__bf16));
  float*  dbcb    = (float*)carve((size_t)MROWS * DBCW * sizeof(float));
  float*  dtbuf   = (float*)carve((size_t)MROWS * IDIM * sizeof(float));
  __bf16* ybuf    = (__bf16*)carve((size_t)MROWS * IDIM * sizeof(__bf16));

  /* one-shot weight conversion (re-run every call; deterministic) */
  cvt_f32_bf16<<<4096, 256, 0, stream>>>(in_proj_w, w_in_b, (long)NLAYERS * 2 * IDIM * HDIM);
  cvt_f32_bf16<<<4096, 256, 0, stream>>>(out_proj_w, w_out_b, (long)NLAYERS * HDIM * IDIM);
  cvt_f32_bf16<<<1024, 256, 0, stream>>>(x_proj_w, w_xp_b, (long)NLAYERS * DBCW * IDIM);

  embed_kernel<<<MROWS, 256, 0, stream>>>(tokens, emb, hbuf, maskb);

  for (int ly = 0; ly < NLAYERS; ly++) {
    rmsnorm_kernel<<<MROWS, 256, 0, stream>>>(hbuf, norm_w + (size_t)ly * HDIM, xb);

    { /* proj = x @ in_proj^T : 8192 x 3072, K=768 (aligned, no guards) */
      dim3 g((2 * IDIM) / TN, MROWS / TM);
      gemm_bf16_wmma<false><<<g, 256, 0, stream>>>(xb, w_in_b + (size_t)ly * 2 * IDIM * HDIM,
                                                   projb, MROWS, 2 * IDIM, HDIM, 0);
    }

    conv_silu_kernel<<<(BB * IDIM + 255) / 256, 256, 0, stream>>>(
        projb, maskb, conv_w + (size_t)ly * IDIM * KCONV, conv_b + (size_t)ly * IDIM, u2f, u2b);

    { /* dbc = u2 @ x_proj^T : 8192 x 80, K=1536 (N bounds-checked) */
      dim3 g((DBCW + TN - 1) / TN, MROWS / TM);
      gemm_bf16_wmma<true><<<g, 256, 0, stream>>>(u2b, w_xp_b + (size_t)ly * DBCW * IDIM,
                                                  dbcb, MROWS, DBCW, IDIM, 0);
    }

    {
      long tot = (long)MROWS * IDIM;
      dt_kernel<<<(unsigned)((tot + 255) / 256), 256, 0, stream>>>(
          dbcb, dt_w + (size_t)ly * IDIM * RRANK, dt_b + (size_t)ly * IDIM, dtbuf);
    }

    scan_kernel<<<(BB * IDIM + 255) / 256, 256, 0, stream>>>(
        dbcb, dtbuf, u2f, projb, A_log + (size_t)ly * IDIM * NSTATE, Dp + (size_t)ly * IDIM, ybuf);

    { /* h += y @ out_proj^T : 8192 x 768, K=1536, accumulate (aligned) */
      dim3 g(HDIM / TN, MROWS / TM);
      gemm_bf16_wmma<false><<<g, 256, 0, stream>>>(ybuf, w_out_b + (size_t)ly * HDIM * IDIM,
                                                   hbuf, MROWS, HDIM, IDIM, 1);
    }
  }

  final_kernel<<<BB, 256, 0, stream>>>(hbuf, norm_f_w, fc_w, fc_b, out);
}
